// MultiHeadAttention_36232344109143
// MI455X (gfx1250) — matrix operand order repaired and checked by reading, not compiled, not run on hardware
//
#include <hip/hip_runtime.h>
#include <math.h>
#include <type_traits>

typedef __attribute__((ext_vector_type(16))) _Float16 v16h;
typedef __attribute__((ext_vector_type(8)))  _Float16 v8h;
typedef __attribute__((ext_vector_type(8)))  float    v8f;

#define D_MODEL 1024
#define SEQ     2048
#define BATCH   2
#define NHEAD   16
#define HDIM    64
#define MTOT    (BATCH*SEQ)   /* 4096 rows */

#define NEG_BIG (-1.0e38f)
#define LOG2E_DIV8 0.180336880f   /* log2(e) / sqrt(64) */

// ---------------- WMMA helpers ----------------

__device__ __forceinline__ v8f wmma_f16(v16h a, v16h b, v8f c) {
  // (neg_a, A, neg_b, B, c_mod, C, reuse_a, reuse_b)
  return __builtin_amdgcn_wmma_f32_16x16x32_f16(false, a, false, b, (short)0, c, false, false);
}

// A-matrix 16x32 f16 fragment: lane holds K = {8*hi..8*hi+7} U {16+8*hi..+7}
__device__ __forceinline__ v16h load_a_frag(const _Float16* rowptr, int kbase, int hi) {
  union { v16h v; v8h h[2]; } u;
  u.h[0] = *(const v8h*)(rowptr + kbase + hi * 8);
  u.h[1] = *(const v8h*)(rowptr + kbase + 16 + hi * 8);
  return u.v;
}

// B-matrix 32x16 f16 fragment: lane (column) holds K = {8*hi..8*hi+7} U {16+8*hi..+7}
__device__ __forceinline__ v16h load_b_frag(const _Float16* colptr, int kbase, int hi) {
  union { v16h v; v8h h[2]; } u;
  u.h[0] = *(const v8h*)(colptr + kbase + hi * 8);
  u.h[1] = *(const v8h*)(colptr + kbase + 16 + hi * 8);
  return u.v;
}

// ---------------- v_permlane16 XOR butterflies (VALU crossbar, no LDS) ----------------

__device__ __forceinline__ float pl16(float v, int sel_lo, int sel_hi) {
  int i = __float_as_int(v);
  i = __builtin_amdgcn_permlane16(i, i, sel_lo, sel_hi, true, false);
  return __int_as_float(i);
}
// reduce across each 16-lane half; result replicated in all 16 lanes
__device__ __forceinline__ float redmax16(float v) {
  v = fmaxf(v, pl16(v, 0x67452301, (int)0xEFCDAB89));   // xor 1
  v = fmaxf(v, pl16(v, 0x54761032, (int)0xDCFE98BA));   // xor 2
  v = fmaxf(v, pl16(v, 0x32107654, (int)0xBA98FEDC));   // xor 4
  v = fmaxf(v, pl16(v, (int)0xFEDCBA98, 0x76543210));   // xor 8
  return v;
}
__device__ __forceinline__ float redsum16(float v) {
  v += pl16(v, 0x67452301, (int)0xEFCDAB89);
  v += pl16(v, 0x54761032, (int)0xDCFE98BA);
  v += pl16(v, 0x32107654, (int)0xBA98FEDC);
  v += pl16(v, (int)0xFEDCBA98, 0x76543210);
  return v;
}

// ---------------- f32 -> f16 convert ----------------

__global__ void cvt_f32_f16(const float* __restrict__ in, _Float16* __restrict__ out, int n) {
  int i = blockIdx.x * 256 + threadIdx.x;
  if (i < n) out[i] = (_Float16)in[i];
}

// ---------------- GEMM: C = A[M,K] @ W[N,K]^T, 16x64 tile per wave ----------------
// MODE 0: f16 out row-major [M,N]
// MODE 1: f16 out transposed to Vt[b*H+h][d][s]   (m=b*S+s, n=h*64+d)
// MODE 2: f32 out row-major + bias (final projection)
template <int MODE>
__global__ void __launch_bounds__(256)
gemm_xwT(const _Float16* __restrict__ A, const _Float16* __restrict__ W,
         _Float16* __restrict__ Ch, float* __restrict__ Cf,
         const float* __restrict__ bias, int M, int N, int K) {
  const int wid  = blockIdx.x * 8 + (threadIdx.x >> 5);
  const int lane = threadIdx.x & 31;
  const int ln = lane & 15, hi = lane >> 4;
  const int ntiles = N >> 6;                    // 64-wide N blocks
  const int mt = wid / ntiles, nt = wid % ntiles;
  const int m0 = mt * 16, n0 = nt * 64;
  if (m0 >= M) return;

  const _Float16* arow = A + (size_t)(m0 + ln) * K;      // A fragment row = M index
  const _Float16* wr[4];
#pragma unroll
  for (int t = 0; t < 4; ++t) wr[t] = W + (size_t)(n0 + t * 16 + ln) * K;

  v8f c[4] = {{}, {}, {}, {}};
  for (int k = 0; k < K; k += 32) {
    v16h a = load_a_frag(arow, k, hi);                   // reused across 4 N-tiles
#pragma unroll
    for (int t = 0; t < 4; ++t)
      c[t] = wmma_f16(a, load_b_frag(wr[t], k, hi), c[t]);
  }

  if (MODE == 0) {
#pragma unroll
    for (int t = 0; t < 4; ++t)
#pragma unroll
      for (int r = 0; r < 8; ++r)
        Ch[(size_t)(m0 + r + 8 * hi) * N + n0 + t * 16 + ln] = (_Float16)c[t][r];
  } else if (MODE == 1) {
#pragma unroll
    for (int t = 0; t < 4; ++t)
#pragma unroll
      for (int r = 0; r < 8; ++r) {
        int m = m0 + r + 8 * hi, n = n0 + t * 16 + ln;
        int b = m >> 11, s = m & (SEQ - 1);
        int h = n >> 6,  d = n & 63;
        Ch[((size_t)(b * NHEAD + h) * HDIM + d) * SEQ + s] = (_Float16)c[t][r];
      }
  } else {
#pragma unroll
    for (int t = 0; t < 4; ++t) {
      float bv = bias[n0 + t * 16 + ln];
#pragma unroll
      for (int r = 0; r < 8; ++r)
        Cf[(size_t)(m0 + r + 8 * hi) * N + n0 + t * 16 + ln] = c[t][r] + bv;
    }
  }
}

// ---------------- Flash attention: one wave per (b,h,q-tile of 16) ----------------

__global__ void __launch_bounds__(256)
attn_kernel(const _Float16* __restrict__ Q, const _Float16* __restrict__ Kh,
            const _Float16* __restrict__ Vt, _Float16* __restrict__ ctx) {
  __shared__ _Float16 plds[8][16 * 32];   // per-wave 16x32 P tile
  const int wslot = threadIdx.x >> 5;
  const int wid   = blockIdx.x * 8 + wslot;
  const int lane  = threadIdx.x & 31;
  const int ln = lane & 15, hi = lane >> 4;

  const int qtiles = SEQ / 16;                 // 128
  const int bh = wid / qtiles;                 // 0..B*H-1
  const int qt = wid % qtiles;
  const int b  = bh / NHEAD, h = bh % NHEAD;
  const int q0 = qt * 16;
  const size_t rowbase = (size_t)b * SEQ;

  // Q fragments for 16x64 tile (two K=32 steps over d)
  const _Float16* qrow = Q + (rowbase + q0 + ln) * D_MODEL + h * HDIM;
  v16h qf0 = load_a_frag(qrow, 0, hi);
  v16h qf1 = load_a_frag(qrow, 32, hi);

  const _Float16* kbase_p = Kh + (rowbase + ln) * D_MODEL + h * HDIM;
  const _Float16* vbase_p = Vt + ((size_t)bh * HDIM + ln) * SEQ;
  _Float16* pl = plds[wslot];

  float mrow[8], lrow[8];
  v8f o[4];
#pragma unroll
  for (int r = 0; r < 8; ++r) { mrow[r] = NEG_BIG; lrow[r] = 0.f; }
#pragma unroll
  for (int t = 0; t < 4; ++t) o[t] = (v8f){};

  auto step = [&](int kb, auto mask_t) {
    constexpr bool MASK = decltype(mask_t)::value;
    // ---- scores for 32 keys: two 16x16 tiles, exp2-domain scale ----
    v8f s[2];
#pragma unroll
    for (int j = 0; j < 2; ++j) {
      const _Float16* krow = kbase_p + (size_t)(kb + j * 16) * D_MODEL;
      v8f acc = {};
      acc = wmma_f16(qf0, load_b_frag(krow, 0, hi), acc);
      acc = wmma_f16(qf1, load_b_frag(krow, 32, hi), acc);
      s[j] = acc;
    }
    float pmx[8];
#pragma unroll
    for (int r = 0; r < 8; ++r) {
      float s0 = s[0][r] * LOG2E_DIV8;
      float s1 = s[1][r] * LOG2E_DIV8;
      if (MASK) {
        int qg = q0 + r + 8 * hi;
        s0 = (kb + ln      <= qg) ? s0 : NEG_BIG;
        s1 = (kb + 16 + ln <= qg) ? s1 : NEG_BIG;
      }
      s[0][r] = s0; s[1][r] = s1;
      pmx[r] = redmax16(fmaxf(s0, s1));        // row max via v_permlane16 butterflies
    }
    // ---- online softmax update (exp2 domain) ----
    float al[8], rsum[8];
#pragma unroll
    for (int r = 0; r < 8; ++r) {
      float mnew = fmaxf(mrow[r], pmx[r]);
      al[r] = exp2f(mrow[r] - mnew);
      mrow[r] = mnew;
      float p0 = exp2f(s[0][r] - mnew);
      float p1 = exp2f(s[1][r] - mnew);
      s[0][r] = p0; s[1][r] = p1;
      rsum[r] = p0 + p1;
    }
#pragma unroll
    for (int r = 0; r < 8; ++r)
      lrow[r] = lrow[r] * al[r] + redsum16(rsum[r]);
#pragma unroll
    for (int t = 0; t < 4; ++t)
#pragma unroll
      for (int r = 0; r < 8; ++r) o[t][r] *= al[r];

    // ---- P (C-layout) -> LDS -> A-fragment layout ----
#pragma unroll
    for (int j = 0; j < 2; ++j)
#pragma unroll
      for (int r = 0; r < 8; ++r)
        pl[(r + 8 * hi) * 32 + j * 16 + ln] = (_Float16)s[j][r];
    asm volatile("s_wait_dscnt 0" ::: "memory");   // same-wave DS ops are in-order
    v16h pf = load_a_frag(pl + ln * 32, 0, hi);    // row q = ln, K = 32 keys

    // ---- O += P @ V  (Vt is [bh][d][s]: contiguous along keys) ----
#pragma unroll
    for (int t = 0; t < 4; ++t)
      o[t] = wmma_f16(pf, load_b_frag(vbase_p + (size_t)t * 16 * SEQ, kb, hi), o[t]);
    asm volatile("s_wait_dscnt 0" ::: "memory");   // reads done before next-iter overwrite
  };

  // Tiles fully below the causal diagonal need no masking: kb+31 <= q0
  const int kend  = q0 + 16;                     // exclusive key bound
  const int nfull = (q0 + 1) / 32;               // count of unmasked 32-key tiles
  int kb = 0;
  for (; kb < nfull * 32; kb += 32) step(kb, std::false_type{});
  for (; kb < kend;       kb += 32) step(kb, std::true_type{});

  // ---- normalize and store ctx[b, q, h*64+d] as f16 ----
  float inv[8];
#pragma unroll
  for (int r = 0; r < 8; ++r) inv[r] = 1.0f / lrow[r];
#pragma unroll
  for (int t = 0; t < 4; ++t)
#pragma unroll
    for (int r = 0; r < 8; ++r)
      ctx[(rowbase + q0 + r + 8 * hi) * D_MODEL + h * HDIM + t * 16 + ln] =
          (_Float16)(o[t][r] * inv[r]);
}

// ---------------- launch ----------------

extern "C" void kernel_launch(void* const* d_in, const int* in_sizes, int n_in,
                              void* d_out, int out_size, void* d_ws, size_t ws_size,
                              hipStream_t stream) {
  const float* x  = (const float*)d_in[0];
  const float* wq = (const float*)d_in[1];
  const float* wk = (const float*)d_in[2];
  const float* wv = (const float*)d_in[3];
  const float* wo = (const float*)d_in[4];
  const float* bo = (const float*)d_in[5];
  float* out = (float*)d_out;

  char* ws = (char*)d_ws;
  const size_t MB = 1u << 20;
  _Float16* x_h   = (_Float16*)(ws + 0 * MB);   // 8 MB [4096,1024]
  _Float16* wq_h  = (_Float16*)(ws + 8 * MB);   // 2 MB
  _Float16* wk_h  = (_Float16*)(ws + 10 * MB);  // 2 MB
  _Float16* wv_h  = (_Float16*)(ws + 12 * MB);  // 2 MB
  _Float16* wo_h  = (_Float16*)(ws + 14 * MB);  // 2 MB
  _Float16* Q_h   = (_Float16*)(ws + 16 * MB);  // 8 MB [4096,1024]
  _Float16* K_h   = (_Float16*)(ws + 24 * MB);  // 8 MB
  _Float16* Vt_h  = (_Float16*)(ws + 32 * MB);  // 8 MB [32][64][2048]
  _Float16* ctx_h = (_Float16*)(ws + 40 * MB);  // 8 MB

  const int NX = MTOT * D_MODEL;                // 4,194,304
  const int NW = D_MODEL * D_MODEL;             // 1,048,576
  cvt_f32_f16<<<NX / 256, 256, 0, stream>>>(x, x_h, NX);
  cvt_f32_f16<<<NW / 256, 256, 0, stream>>>(wq, wq_h, NW);
  cvt_f32_f16<<<NW / 256, 256, 0, stream>>>(wk, wk_h, NW);
  cvt_f32_f16<<<NW / 256, 256, 0, stream>>>(wv, wv_h, NW);
  cvt_f32_f16<<<NW / 256, 256, 0, stream>>>(wo, wo_h, NW);

  const int gemm_blocks = (MTOT / 16) * (D_MODEL / 64) / 8;  // 512
  gemm_xwT<0><<<gemm_blocks, 256, 0, stream>>>(x_h, wq_h, Q_h,  nullptr, nullptr, MTOT, D_MODEL, D_MODEL);
  gemm_xwT<0><<<gemm_blocks, 256, 0, stream>>>(x_h, wk_h, K_h,  nullptr, nullptr, MTOT, D_MODEL, D_MODEL);
  gemm_xwT<1><<<gemm_blocks, 256, 0, stream>>>(x_h, wv_h, Vt_h, nullptr, nullptr, MTOT, D_MODEL, D_MODEL);

  const int attn_blocks = (BATCH * NHEAD * (SEQ / 16)) / 8;  // 512
  attn_kernel<<<attn_blocks, 256, 0, stream>>>(Q_h, K_h, Vt_h, ctx_h);

  gemm_xwT<2><<<gemm_blocks, 256, 0, stream>>>(ctx_h, wo_h, nullptr, out, bo, MTOT, D_MODEL, D_MODEL);
}
